// LayerStack_79096117723303
// MI455X (gfx1250) — matrix operand order repaired and checked by reading, not compiled, not run on hardware
//
#include <hip/hip_runtime.h>
#include <hip/hip_bf16.h>

#define INPUT_SZ 2048
#define H1 256
#define H2 64
#define NB 8
#define BATCH 16384

#define KCHUNK 1024
#define XSTRIDE (KCHUNK + 8)   // f16 elems; 516 dwords % 64 = 4 -> conflict-free
#define H1STRIDE 264           // f16 elems; 132 dwords % 64 = 4; 528B is 16B-aligned
#define H2STRIDE 68            // f32 elems

typedef __attribute__((ext_vector_type(16))) _Float16 v16h;
typedef __attribute__((ext_vector_type(8)))  _Float16 v8h;
typedef __attribute__((ext_vector_type(8)))  float    v8f;

__device__ __forceinline__ v16h ld16(const _Float16* p0, const _Float16* p1) {
    union { v16h v; v8h h[2]; } u;
    u.h[0] = *(const v8h*)p0;   // 16B chunk -> ds_load_b128 / global_load_b128
    u.h[1] = *(const v8h*)p1;
    return u.v;
}

__device__ __forceinline__ float clip01(float v) {
    return fminf(fmaxf(v, 0.0f), 1.0f);
}

// ---------------- prep kernels ----------------

__global__ void zero_counts(int* counts) {
    if (threadIdx.x < NB) counts[threadIdx.x] = 0;
}

__global__ void bin_kernel(const int* __restrict__ bidx, int* counts, int* bins) {
    int i = blockIdx.x * blockDim.x + threadIdx.x;
    if (i < BATCH) {
        int b = bidx[i];
        int pos = atomicAdd(&counts[b], 1);
        bins[b * BATCH + pos] = i;
    }
}

__global__ void cvt_kernel(const float* __restrict__ W1, const float* __restrict__ W2,
                           _Float16* __restrict__ W1h, _Float16* __restrict__ W2h) {
    const int N1 = H1 * NB * INPUT_SZ;   // 4,194,304
    const int N2 = H2 * NB * H1;         // 131,072
    int idx = (blockIdx.x * blockDim.x + threadIdx.x) * 4;
    if (idx < N1) {
        float4 v = *(const float4*)(&W1[idx]);
        W1h[idx + 0] = (_Float16)v.x; W1h[idx + 1] = (_Float16)v.y;
        W1h[idx + 2] = (_Float16)v.z; W1h[idx + 3] = (_Float16)v.w;
    } else {
        int j = idx - N1;
        if (j < N2) {
            float4 v = *(const float4*)(&W2[j]);
            W2h[j + 0] = (_Float16)v.x; W2h[j + 1] = (_Float16)v.y;
            W2h[j + 2] = (_Float16)v.z; W2h[j + 3] = (_Float16)v.w;
        }
    }
}

// ---------------- fused bucketed 3-layer MLP ----------------
// grid: (BATCH/16, NB), block: 256 (8 waves of 32)
__global__ __launch_bounds__(256)
void moe_main(const float* __restrict__ x,
              const float* __restrict__ b1, const float* __restrict__ b2,
              const float* __restrict__ W3, const float* __restrict__ b3,
              const _Float16* __restrict__ W1h, const _Float16* __restrict__ W2h,
              const int* __restrict__ counts, const int* __restrict__ bins,
              float* __restrict__ out) {
    __shared__ _Float16 xlds[16 * XSTRIDE];    // 33,024 B
    __shared__ _Float16 h1lds[16 * H1STRIDE];  //  8,448 B
    __shared__ float    h2lds[16 * H2STRIDE];  //  4,352 B
    __shared__ int      s_idx[16];

    const int bucket = blockIdx.y;
    const int cnt    = counts[bucket];
    const int start  = blockIdx.x * 16;
    if (start >= cnt) return;   // uniform per block

    const int tid    = threadIdx.x;
    const int wave   = tid >> 5;
    const int lane   = tid & 31;
    const int laneN  = lane & 15;
    const int laneHi = lane >> 4;     // 0 or 1

    if (tid < 16) {
        int g = start + tid;
        s_idx[tid] = bins[bucket * BATCH + (g < cnt ? g : start)]; // pad with first valid
    }
    __syncthreads();

    // ---- layer 1: (16 x 2048) @ (2048 x 256)^T, wave w owns output cols [32w,32w+32) ----
    v8f acc0 = {};
    v8f acc1 = {};
    const int o0 = wave * 32;
    const long wrow0 = (long)(bucket * H1 + o0 + laneN) * INPUT_SZ;
    const long wrow1 = wrow0 + (long)16 * INPUT_SZ;

    for (int ch = 0; ch < INPUT_SZ / KCHUNK; ++ch) {
        // stage 16 x KCHUNK fp32 -> f16 LDS (each thread: 16 x float4, coalesced)
        #pragma unroll
        for (int it = 0; it < 16; ++it) {
            int e = it * 1024 + tid * 4;
            int r = e >> 10;
            int c = e & 1023;
            float4 v = *(const float4*)(&x[(size_t)s_idx[r] * INPUT_SZ + ch * KCHUNK + c]);
            _Float16* dst = &xlds[r * XSTRIDE + c];
            dst[0] = (_Float16)v.x; dst[1] = (_Float16)v.y;
            dst[2] = (_Float16)v.z; dst[3] = (_Float16)v.w;
        }
        __syncthreads();

        const _Float16* arow = &xlds[laneN * XSTRIDE];
        #pragma unroll 2
        for (int kk = 0; kk < KCHUNK; kk += 32) {
            const int kA = kk + laneHi * 16;           // A: 16 contiguous K of row m
            const int kB = ch * KCHUNK + kk + laneHi * 16; // B: 16 contiguous K of col n
            v16h a  = ld16(arow + kA, arow + kA + 8);
            v16h bA = ld16(&W1h[wrow0 + kB], &W1h[wrow0 + kB + 8]);
            v16h bB = ld16(&W1h[wrow1 + kB], &W1h[wrow1 + kB + 8]);
            acc0 = __builtin_amdgcn_wmma_f32_16x16x32_f16(false, a, false, bA,
                                                          (short)0, acc0, false, false);
            acc1 = __builtin_amdgcn_wmma_f32_16x16x32_f16(false, a, false, bB,
                                                          (short)0, acc1, false, false);
        }
        __syncthreads();
    }

    // bias + clip(0,1), h1 -> LDS (f16). D layout: (M = r + 8*laneHi, N = laneN)
    {
        const float bias0 = b1[bucket * H1 + o0 + laneN];
        const float bias1 = b1[bucket * H1 + o0 + 16 + laneN];
        #pragma unroll
        for (int r = 0; r < 8; ++r) {
            int M = r + laneHi * 8;
            h1lds[M * H1STRIDE + o0 + laneN]      = (_Float16)clip01(acc0[r] + bias0);
            h1lds[M * H1STRIDE + o0 + 16 + laneN] = (_Float16)clip01(acc1[r] + bias1);
        }
    }
    __syncthreads();

    // ---- layer 2: (16 x 256) @ (256 x 64)^T, waves 0..3 own 16 cols each ----
    if (wave < 4) {
        v8f acc = {};
        const int o2 = wave * 16;
        const long w2row = (long)(bucket * H2 + o2 + laneN) * H1;
        const _Float16* arow = &h1lds[laneN * H1STRIDE];
        #pragma unroll
        for (int kk = 0; kk < H1; kk += 32) {
            const int kA = kk + laneHi * 16;
            const int kB = kk + laneHi * 16;
            v16h a = ld16(arow + kA, arow + kA + 8);
            v16h b = ld16(&W2h[w2row + kB], &W2h[w2row + kB + 8]);
            acc = __builtin_amdgcn_wmma_f32_16x16x32_f16(false, a, false, b,
                                                         (short)0, acc, false, false);
        }
        const float bias = b2[bucket * H2 + o2 + laneN];
        #pragma unroll
        for (int r = 0; r < 8; ++r) {
            int M = r + laneHi * 8;
            h2lds[M * H2STRIDE + o2 + laneN] = clip01(acc[r] + bias);
        }
    }
    __syncthreads();

    // ---- layer 3: (16 x 64) @ (64 x 1) + b3, masked scatter ----
    if (tid < 16) {
        const float* w3 = &W3[bucket * H2];
        float sum = b3[bucket];
        #pragma unroll
        for (int k = 0; k < H2; ++k) sum += h2lds[tid * H2STRIDE + k] * w3[k];
        if (start + tid < cnt) out[s_idx[tid]] = sum;
    }
}

// ---------------- launcher ----------------

extern "C" void kernel_launch(void* const* d_in, const int* in_sizes, int n_in,
                              void* d_out, int out_size, void* d_ws, size_t ws_size,
                              hipStream_t stream) {
    const float* x    = (const float*)d_in[0];
    const int*   bidx = (const int*)  d_in[1];
    const float* W1   = (const float*)d_in[2];
    const float* b1   = (const float*)d_in[3];
    const float* W2   = (const float*)d_in[4];
    const float* b2   = (const float*)d_in[5];
    const float* W3   = (const float*)d_in[6];
    const float* b3   = (const float*)d_in[7];
    float* out = (float*)d_out;

    // workspace layout (~8.9 MB total)
    char* ws = (char*)d_ws;
    _Float16* W1h   = (_Float16*)ws;                               // 8,388,608 B
    _Float16* W2h   = (_Float16*)(ws + 8 * 1024 * 1024);           //   262,144 B
    int*      counts = (int*)(ws + 8 * 1024 * 1024 + 512 * 1024);  //        32 B
    int*      bins   = (int*)(ws + 8 * 1024 * 1024 + 512 * 1024 + 256); // 524,288 B

    zero_counts<<<1, 32, 0, stream>>>(counts);
    bin_kernel<<<BATCH / 256, 256, 0, stream>>>(bidx, counts, bins);

    const int ncvt = (H1 * NB * INPUT_SZ + H2 * NB * H1) / 4; // elems/4 per thread
    cvt_kernel<<<(ncvt + 255) / 256, 256, 0, stream>>>(W1, W2, W1h, W2h);

    dim3 grid(BATCH / 16, NB);
    moe_main<<<grid, 256, 0, stream>>>(x, b1, b2, W3, b3, W1h, W2h, counts, bins, out);
}
